// MambaEncoder_82016695484683
// MI455X (gfx1250) — matrix operand order repaired and checked by reading, not compiled, not run on hardware
//
#include <hip/hip_runtime.h>
#include <hip/hip_bf16.h>

// ---- problem constants (from reference) ----
#define L_SEQ   2048
#define DM      1024     // D_MODEL
#define DI      2048     // D_INNER
#define DS      16       // D_STATE
#define DTR     64       // DT_RANK
#define DCONV   4
#define NXP     96       // DT_RANK + 2*D_STATE
#define EPS_C   1e-5f

typedef __attribute__((ext_vector_type(16))) _Float16 v16h;
typedef __attribute__((ext_vector_type(8)))  float    v8f;

// ---------------------------------------------------------------------------
// Stage weights: W (N x K, f32 row-major) -> Wt (K x N, f16 row-major).
// LDS-tiled so both the read (along K) and the write (along N) are coalesced.
// Wt is the GEMM B operand: lane l holds column n=l&15, 16 k values
// strided by N (16-bit loads, not b128) in the GEMM.
// ---------------------------------------------------------------------------
__global__ void k_convert_transpose(const float* __restrict__ W,
                                    _Float16* __restrict__ Wt,
                                    int N, int K) {
  __shared__ _Float16 tile[16][17];
  int k0 = blockIdx.x * 16;
  int n0 = blockIdx.y * 16;
  int tx = threadIdx.x & 15;
  int ty = threadIdx.x >> 4;   // 256 threads -> full 16x16 tile
  tile[ty][tx] = (_Float16)W[(size_t)(n0 + ty) * K + (k0 + tx)];
  __syncthreads();
  Wt[(size_t)(k0 + ty) * N + (n0 + tx)] = tile[tx][ty];
}

// ---------------------------------------------------------------------------
// resid_out = h (+ resid_in);  norm = rmsnorm(resid_out) * w
// norm written as f16 (GEMM A operand) and/or f32 (final output).
// One block (256 threads) per token row of 1024.
// ---------------------------------------------------------------------------
__global__ void k_add_rmsnorm(const float* __restrict__ h,
                              const float* __restrict__ resid_in,
                              float* __restrict__ resid_out,
                              _Float16* __restrict__ norm_h,
                              float* __restrict__ norm_f,
                              const float* __restrict__ w) {
  __shared__ float red[256];
  int row = blockIdx.x;
  const float* hr = h + (size_t)row * DM;
  const float* rr = resid_in ? resid_in + (size_t)row * DM : nullptr;
  float v[4];
  float ss = 0.f;
#pragma unroll
  for (int j = 0; j < 4; ++j) {
    int c = threadIdx.x + j * 256;
    float x = hr[c] + (rr ? rr[c] : 0.f);
    v[j] = x;
    ss += x * x;
  }
  red[threadIdx.x] = ss;
  __syncthreads();
  for (int s = 128; s > 0; s >>= 1) {
    if (threadIdx.x < s) red[threadIdx.x] += red[threadIdx.x + s];
    __syncthreads();
  }
  float inv = rsqrtf(red[0] / (float)DM + EPS_C);
#pragma unroll
  for (int j = 0; j < 4; ++j) {
    int c = threadIdx.x + j * 256;
    float xn = v[j] * inv * w[c];
    resid_out[(size_t)row * DM + c] = v[j];
    if (norm_h) norm_h[(size_t)row * DM + c] = (_Float16)xn;
    if (norm_f) norm_f[(size_t)row * DM + c] = xn;
  }
}

// ---------------------------------------------------------------------------
// WMMA GEMM: C[M,N] (f32) = A[M,K] (f16 row-major) * Bt[K,N] (f16 row-major)
// One wave32 computes a 32x32 output block: 2x2 tiles of 16x16, 4 WMMAs per
// 32-deep k-step sharing 2 A-fragments and 2 B-fragments (A rows b128-loadable, B strided by N).
//
// Fragment layouts per ISA:
//  A: lane l -> row m=l&15; e<8: k=kb+8*(l>>4)+e ; e>=8: k=kb+16+8*(l>>4)+e-8
//  B: lane l -> column n=l&15; element e -> same k as A   (strided by N in Bt)
//  D: element r -> m=r+8*(l>>4), n=l&15
// ---------------------------------------------------------------------------
__global__ void __launch_bounds__(256)
k_gemm_wmma_f16(const _Float16* __restrict__ A,
                const _Float16* __restrict__ Bt,
                float* __restrict__ C,
                int M, int N, int K) {
  int wave = (blockIdx.x * blockDim.x + threadIdx.x) >> 5;
  int lane = threadIdx.x & 31;
  int tilesN = N >> 5;                       // 32-wide N blocks
  int tM = wave / tilesN;
  int tN = wave % tilesN;
  if (tM * 32 >= M) return;                  // wave-uniform guard
  int halfSel = lane >> 4;
  int mr = lane & 15;
  const _Float16* Arow0 = A + (size_t)(tM * 32 + mr) * K;
  const _Float16* Arow1 = Arow0 + (size_t)16 * K;
  const _Float16* Brow  = Bt + (size_t)tN * 32 + mr;
  v8f acc00 = {}, acc01 = {}, acc10 = {}, acc11 = {};
  for (int kb = 0; kb < K; kb += 32) {
    const _Float16* pa0 = Arow0 + kb + 8 * halfSel;
    const _Float16* pa1 = Arow1 + kb + 8 * halfSel;
    const _Float16* pb  = Brow + (size_t)(kb + 8 * halfSel) * N;
    if (kb + 32 < K) __builtin_prefetch(pb + (size_t)32 * N, 0, 1);
    v16h a0, a1, b0, b1;
#pragma unroll
    for (int e = 0; e < 8; ++e) {
      a0[e] = pa0[e];  a0[8 + e] = pa0[16 + e];
      a1[e] = pa1[e];  a1[8 + e] = pa1[16 + e];
    }
#pragma unroll
    for (int e = 0; e < 8; ++e) {
      b0[e] = pb[(size_t)e * N];  b0[8 + e] = pb[(size_t)(16 + e) * N];
      b1[e] = pb[(size_t)e * N + 16];  b1[8 + e] = pb[(size_t)(16 + e) * N + 16];
    }
    acc00 = __builtin_amdgcn_wmma_f32_16x16x32_f16(false, a0, false, b0, (short)0, acc00, false, false);
    acc01 = __builtin_amdgcn_wmma_f32_16x16x32_f16(false, a0, false, b1, (short)0, acc01, false, false);
    acc10 = __builtin_amdgcn_wmma_f32_16x16x32_f16(false, a1, false, b0, (short)0, acc10, false, false);
    acc11 = __builtin_amdgcn_wmma_f32_16x16x32_f16(false, a1, false, b1, (short)0, acc11, false, false);
  }
  int nc = lane & 15;
#pragma unroll
  for (int r = 0; r < 8; ++r) {
    int m0 = tM * 32 + r + 8 * halfSel;
    int n0 = tN * 32 + nc;
    C[(size_t)m0 * N + n0]             = acc00[r];
    C[(size_t)m0 * N + n0 + 16]        = acc01[r];
    C[(size_t)(m0 + 16) * N + n0]      = acc10[r];
    C[(size_t)(m0 + 16) * N + n0 + 16] = acc11[r];
  }
}

// ---------------------------------------------------------------------------
// causal depthwise conv (kernel 4) + SiLU on xi = xz[:, :DI]; writes f32 + f16
// ---------------------------------------------------------------------------
__global__ void k_conv_silu(const float* __restrict__ xz,
                            const float* __restrict__ cw,
                            const float* __restrict__ cb,
                            float* __restrict__ xc,
                            _Float16* __restrict__ xc_h) {
  int idx = blockIdx.x * blockDim.x + threadIdx.x;
  if (idx >= L_SEQ * DI) return;
  int l = idx / DI, d = idx % DI;
  float acc = cb[d];
#pragma unroll
  for (int k = 0; k < DCONV; ++k) {
    int ls = l - (DCONV - 1) + k;
    if (ls >= 0) acc += xz[(size_t)ls * (2 * DI) + d] * cw[d * DCONV + k];
  }
  float s = acc / (1.f + expf(-acc));        // SiLU
  xc[idx] = s;
  xc_h[idx] = (_Float16)s;
}

// dt_low = dbc[:, :DTR] -> f16 for dt GEMM A operand
__global__ void k_slice_dtlow(const float* __restrict__ dbc,
                              _Float16* __restrict__ dtlow_h) {
  int idx = blockIdx.x * blockDim.x + threadIdx.x;
  if (idx >= L_SEQ * DTR) return;
  int l = idx / DTR, r = idx % DTR;
  dtlow_h[idx] = (_Float16)dbc[(size_t)l * NXP + r];
}

// dt = softplus(dt_pre + dt_b)  (in place)
__global__ void k_softplus_bias(float* __restrict__ dt,
                                const float* __restrict__ b) {
  int idx = blockIdx.x * blockDim.x + threadIdx.x;
  if (idx >= L_SEQ * DI) return;
  float x = dt[idx] + b[idx % DI];
  dt[idx] = (x > 20.f) ? x : log1pf(expf(x));
}

// ---------------------------------------------------------------------------
// selective scan: one thread per channel d (2048 lanes = 64 waves),
// 16 states in registers, B/C broadcast per timestep via LDS.
// ---------------------------------------------------------------------------
__global__ void k_scan(const float* __restrict__ xc,
                       const float* __restrict__ dt,
                       const float* __restrict__ dbc,
                       const float* __restrict__ A_log,
                       const float* __restrict__ Dp,
                       float* __restrict__ y) {
  int d = blockIdx.x * blockDim.x + threadIdx.x;   // 0..DI-1
  float Av[DS], hst[DS];
#pragma unroll
  for (int n = 0; n < DS; ++n) {
    Av[n] = -expf(A_log[(size_t)d * DS + n]);
    hst[n] = 0.f;
  }
  float Dd = Dp[d];
  __shared__ float Bs[DS], Cs[DS];
  for (int l = 0; l < L_SEQ; ++l) {
    if (threadIdx.x < 32) {
      float vv = dbc[(size_t)l * NXP + DTR + threadIdx.x];
      if (threadIdx.x < DS) Bs[threadIdx.x] = vv;
      else                  Cs[threadIdx.x - DS] = vv;
    }
    __syncthreads();
    float dtv = dt[(size_t)l * DI + d];
    float u   = xc[(size_t)l * DI + d];
    float du  = dtv * u;
    float yv  = 0.f;
#pragma unroll
    for (int n = 0; n < DS; ++n) {
      hst[n] = __expf(dtv * Av[n]) * hst[n] + du * Bs[n];
      yv += hst[n] * Cs[n];
    }
    y[(size_t)l * DI + d] = yv + u * Dd;
    __syncthreads();
  }
}

// ym = y * silu(z), written as f16 (out_proj GEMM A operand)
__global__ void k_gate(const float* __restrict__ y,
                       const float* __restrict__ xz,
                       _Float16* __restrict__ ym_h) {
  int idx = blockIdx.x * blockDim.x + threadIdx.x;
  if (idx >= L_SEQ * DI) return;
  int l = idx / DI, d = idx % DI;
  float z = xz[(size_t)l * (2 * DI) + DI + d];
  float g = z / (1.f + expf(-z));
  ym_h[idx] = (_Float16)(y[idx] * g);
}

// ---------------------------------------------------------------------------
static inline void launch_gemm(const _Float16* A, const _Float16* Bt, float* C,
                               int M, int N, int K, hipStream_t s) {
  int waves = (M / 32) * (N / 32);
  int blocks = (waves + 7) / 8;              // 8 waves (256 threads) per block
  k_gemm_wmma_f16<<<blocks, 256, 0, s>>>(A, Bt, C, M, N, K);
}

static inline void launch_transpose(const float* W, _Float16* Wt,
                                    int N, int K, hipStream_t s) {
  dim3 grid(K / 16, N / 16);
  k_convert_transpose<<<grid, 256, 0, s>>>(W, Wt, N, K);
}

extern "C" void kernel_launch(void* const* d_in, const int* in_sizes, int n_in,
                              void* d_out, int out_size, void* d_ws, size_t ws_size,
                              hipStream_t stream) {
  const float* hidden  = (const float*)d_in[0];   // (1,2048,1024)
  const float* norm_w  = (const float*)d_in[1];   // (2,1024)
  const float* in_w    = (const float*)d_in[2];   // (2,4096,1024)
  const float* conv_w  = (const float*)d_in[3];   // (2,2048,4)
  const float* conv_b  = (const float*)d_in[4];   // (2,2048)
  const float* xp_w    = (const float*)d_in[5];   // (2,96,2048)
  const float* dt_w    = (const float*)d_in[6];   // (2,2048,64)
  const float* dt_b    = (const float*)d_in[7];   // (2,2048)
  const float* A_log   = (const float*)d_in[8];   // (2,2048,16)
  const float* Dp      = (const float*)d_in[9];   // (2,2048)
  const float* out_w   = (const float*)d_in[10];  // (2,1024,2048)
  const float* norm_fw = (const float*)d_in[11];  // (1024,)
  float* out = (float*)d_out;

  // ---- workspace carve-up (all chunks 16B aligned) ----
  char* ws = (char*)d_ws;
  auto allocF = [&](size_t n) { float* p = (float*)ws; ws += n * sizeof(float); return p; };
  auto allocH = [&](size_t n) { _Float16* p = (_Float16*)ws; ws += n * sizeof(_Float16); return p; };

  float* resid = allocF((size_t)L_SEQ * DM);
  float* xz    = allocF((size_t)L_SEQ * 2 * DI);
  float* xc    = allocF((size_t)L_SEQ * DI);
  float* dbc   = allocF((size_t)L_SEQ * NXP);
  float* dt    = allocF((size_t)L_SEQ * DI);
  float* yb    = allocF((size_t)L_SEQ * DI);
  float* hbuf  = allocF((size_t)L_SEQ * DM);
  _Float16* xnorm_h = allocH((size_t)L_SEQ * DM);
  _Float16* xc_h    = allocH((size_t)L_SEQ * DI);
  _Float16* dtlow_h = allocH((size_t)L_SEQ * DTR);
  _Float16* ym_h    = allocH((size_t)L_SEQ * DI);
  _Float16* w_in_t  = allocH((size_t)DM * 2 * DI);     // 1024 x 4096  (KxN)
  _Float16* w_xp_t  = allocH((size_t)DI * NXP);        // 2048 x 96
  _Float16* w_dt_t  = allocH((size_t)DTR * DI);        // 64 x 2048
  _Float16* w_out_t = allocH((size_t)DI * DM);         // 2048 x 1024

  const float* h_cur = hidden;
  for (int layer = 0; layer < 2; ++layer) {
    // residual accumulate + rmsnorm (norm -> f16 for GEMM)
    k_add_rmsnorm<<<L_SEQ, 256, 0, stream>>>(
        h_cur, layer == 0 ? nullptr : resid, resid, xnorm_h, nullptr,
        norm_w + (size_t)layer * DM);

    // stage this layer's weights: f32 (N x K) -> f16 transposed (K x N)
    launch_transpose(in_w  + (size_t)layer * 2 * DI * DM, w_in_t,  2 * DI, DM,  stream);
    launch_transpose(xp_w  + (size_t)layer * NXP * DI,    w_xp_t,  NXP,    DI,  stream);
    launch_transpose(dt_w  + (size_t)layer * DI * DTR,    w_dt_t,  DI,     DTR, stream);
    launch_transpose(out_w + (size_t)layer * DM * DI,     w_out_t, DM,     DI,  stream);

    // in_proj: xz[L,4096] = xnorm[L,1024] @ in_w^T
    launch_gemm(xnorm_h, w_in_t, xz, L_SEQ, 2 * DI, DM, stream);

    // depthwise conv + SiLU
    k_conv_silu<<<(L_SEQ * DI) / 256, 256, 0, stream>>>(
        xz, conv_w + (size_t)layer * DI * DCONV, conv_b + (size_t)layer * DI, xc, xc_h);

    // x_proj: dbc[L,96] = xc[L,2048] @ xp_w^T
    launch_gemm(xc_h, w_xp_t, dbc, L_SEQ, NXP, DI, stream);

    // dt projection: dt[L,2048] = dt_low[L,64] @ dt_w^T ; then softplus(+bias)
    k_slice_dtlow<<<(L_SEQ * DTR) / 256, 256, 0, stream>>>(dbc, dtlow_h);
    launch_gemm(dtlow_h, w_dt_t, dt, L_SEQ, DI, DTR, stream);
    k_softplus_bias<<<(L_SEQ * DI) / 256, 256, 0, stream>>>(dt, dt_b + (size_t)layer * DI);

    // selective scan (sequential over L, parallel over channels)
    k_scan<<<DI / 256, 256, 0, stream>>>(
        xc, dt, dbc, A_log + (size_t)layer * DI * DS, Dp + (size_t)layer * DI, yb);

    // gating, then out_proj: hbuf[L,1024] = (y*silu(z))[L,2048] @ out_w^T
    k_gate<<<(L_SEQ * DI) / 256, 256, 0, stream>>>(yb, xz, ym_h);
    launch_gemm(ym_h, w_out_t, hbuf, L_SEQ, DM, DI, stream);

    h_cur = hbuf;
  }

  // final: out = rmsnorm(hbuf + resid) * norm_f_w   (resid scratch -> xz region)
  k_add_rmsnorm<<<L_SEQ, 256, 0, stream>>>(hbuf, resid, xz, nullptr, out, norm_fw);
}
